// GroupedMoEWrapper_72636486910164
// MI455X (gfx1250) — matrix operand order repaired and checked by reading, not compiled, not run on hardware
//
#include <hip/hip_runtime.h>
#include <hip/hip_bf16.h>

// ---------------------------------------------------------------------------
// MoE (top-2 of 8 routed experts, SwiGLU) + shared SwiGLU expert, MI455X.
// bf16 WMMA (v_wmma_f32_16x16x32_bf16) with fp32 accumulation.
// Tile staging uses CDNA5 async global->LDS DMA (ASYNCcnt) to bypass VGPRs.
// ---------------------------------------------------------------------------

typedef __attribute__((ext_vector_type(16))) __bf16 v16bf;
typedef __attribute__((ext_vector_type(8)))  __bf16 bf16x8;
typedef __attribute__((ext_vector_type(8)))  float  v8f;

constexpr int T_TOK = 4096;   // B*S
constexpr int DIM   = 2048;   // D
constexpr int NEXP  = 8;
constexpr int HDIM  = 1024;   // routed expert ffn dim
constexpr int HSDIM = 2048;   // shared expert ffn dim

// -------------------------------- helpers ----------------------------------

__device__ __forceinline__ v8f wmma_bf16(v16bf a, v16bf b, v8f c) {
  // (neg_a, A, neg_b, B, c_mod, C, reuse_a, reuse_b)
  return __builtin_amdgcn_wmma_f32_16x16x32_bf16(false, a, false, b,
                                                 (short)0, c, false, false);
}

// Per-lane async DMA: 16B global -> LDS, tracked by ASYNCcnt (no VGPR data).
__device__ __forceinline__ void async_copy_b128(unsigned lds_addr,
                                                const void* gptr) {
  asm volatile("global_load_async_to_lds_b128 %0, %1, off"
               :
               : "v"(lds_addr), "v"((unsigned long long)(uintptr_t)gptr)
               : "memory");
}
__device__ __forceinline__ void wait_async0() {
  asm volatile("s_wait_asynccnt 0x0" ::: "memory");
}
__device__ __forceinline__ unsigned lds_lo32(const void* p) {
  // LDS aperture: generic addr[31:0] == wave-relative LDS byte address.
  return (unsigned)(uintptr_t)p;
}

// A-fragment: two 16B chunks at +0 and +16 halves (K {kh*8..kh*8+7, 16+kh*8..})
__device__ __forceinline__ v16bf ld_frag_split(const __bf16* p) {
  bf16x8 lo = *reinterpret_cast<const bf16x8*>(p);
  bf16x8 hi = *reinterpret_cast<const bf16x8*>(p + 16);
  return __builtin_shufflevector(lo, hi, 0, 1, 2, 3, 4, 5, 6, 7,
                                 8, 9, 10, 11, 12, 13, 14, 15);
}

// B-fragment: 32B contiguous (16 consecutive K for this half-wave's column)
__device__ __forceinline__ v16bf ld_frag_contig(const __bf16* p) {
  bf16x8 lo = *reinterpret_cast<const bf16x8*>(p);
  bf16x8 hi = *reinterpret_cast<const bf16x8*>(p + 8);
  return __builtin_shufflevector(lo, hi, 0, 1, 2, 3, 4, 5, 6, 7,
                                 8, 9, 10, 11, 12, 13, 14, 15);
}

__device__ __forceinline__ float silu(float x) {
  // x * sigmoid(x); v_rcp_f32 fast path instead of IEEE division.
  return x * __builtin_amdgcn_rcpf(1.0f + __expf(-x));
}

// ------------------------------ conversions --------------------------------

__global__ __launch_bounds__(256) void cvt_bf16_kernel(
    const float* __restrict__ in, __bf16* __restrict__ out, long long n) {
  long long i = ((long long)blockIdx.x * blockDim.x + threadIdx.x) * 4;
  if (i < n) {
    float4 v = *reinterpret_cast<const float4*>(in + i);
    out[i + 0] = (__bf16)v.x;
    out[i + 1] = (__bf16)v.y;
    out[i + 2] = (__bf16)v.z;
    out[i + 3] = (__bf16)v.w;
  }
}

// in: [K, N] fp32 (batched over z)  ->  out: [N, K] bf16
__global__ __launch_bounds__(256) void transpose_cvt_kernel(
    const float* __restrict__ in, __bf16* __restrict__ out, int K, int N) {
  __shared__ float tile[32][33];
  const float* src = in + (size_t)blockIdx.z * K * N;
  __bf16* dst = out + (size_t)blockIdx.z * K * N;
  const int nx = blockIdx.x * 32, ky = blockIdx.y * 32;
#pragma unroll
  for (int i = 0; i < 32; i += 8)
    tile[threadIdx.y + i][threadIdx.x] =
        src[(size_t)(ky + threadIdx.y + i) * N + nx + threadIdx.x];
  __syncthreads();
#pragma unroll
  for (int i = 0; i < 32; i += 8)
    dst[(size_t)(nx + threadIdx.y + i) * K + ky + threadIdx.x] =
        (__bf16)tile[threadIdx.x][threadIdx.y + i];
}

// -------------------------------- router -----------------------------------
// Dense combine weights [T, E]: zeros except renormalized top-2 softmax probs.

__global__ __launch_bounds__(256) void router_kernel(
    const float* __restrict__ x, const float* __restrict__ rw,
    float* __restrict__ comb) {
  __shared__ float red[256][NEXP];
  const int t = blockIdx.x, tid = threadIdx.x;
  float acc[NEXP];
#pragma unroll
  for (int e = 0; e < NEXP; ++e) acc[e] = 0.0f;
  for (int j = tid; j < DIM; j += 256) {
    float xv = x[(size_t)t * DIM + j];
#pragma unroll
    for (int e = 0; e < NEXP; ++e) acc[e] += xv * rw[j * NEXP + e];
  }
#pragma unroll
  for (int e = 0; e < NEXP; ++e) red[tid][e] = acc[e];
  __syncthreads();
  for (int off = 128; off > 0; off >>= 1) {
    if (tid < off) {
#pragma unroll
      for (int e = 0; e < NEXP; ++e) red[tid][e] += red[tid + off][e];
    }
    __syncthreads();
  }
  if (tid == 0) {
    float l[NEXP], p[NEXP];
    float mx = red[0][0];
#pragma unroll
    for (int e = 0; e < NEXP; ++e) { l[e] = red[0][e]; mx = fmaxf(mx, l[e]); }
#pragma unroll
    for (int e = 0; e < NEXP; ++e) p[e] = __expf(l[e] - mx);
    int i1 = 0;
#pragma unroll
    for (int e = 1; e < NEXP; ++e) if (p[e] > p[i1]) i1 = e;
    int i2 = (i1 == 0) ? 1 : 0;
#pragma unroll
    for (int e = 0; e < NEXP; ++e)
      if (e != i1 && p[e] > p[i2]) i2 = e;
    float rs = __builtin_amdgcn_rcpf(p[i1] + p[i2]);
#pragma unroll
    for (int e = 0; e < NEXP; ++e)
      comb[(size_t)t * NEXP + e] =
          (e == i1) ? p[i1] * rs : ((e == i2) ? p[i2] * rs : 0.0f);
  }
}

// -------------------------- GEMM1: X@Wg, X@Wu, SwiGLU ----------------------
// X:[T,K] bf16 row-major; Wgt/Wut:[N,K] bf16 (pre-transposed);
// Hact:[T,N] bf16 = silu(X@Wg) * (X@Wu).  Tile 64x64, 8 waves, K-step 32.

constexpr int BM = 64, BN = 64, LDT = 40;  // LDT halves = 80B row (16B aligned)

__global__ __launch_bounds__(256) void gemm1_swiglu_kernel(
    const __bf16* __restrict__ X, const __bf16* __restrict__ Wgt,
    const __bf16* __restrict__ Wut, __bf16* __restrict__ Hact,
    int Kdim, int Ndim) {
  __shared__ __bf16 As[2][BM * LDT];
  __shared__ __bf16 Bg[2][BN * LDT];
  __shared__ __bf16 Bu[2][BN * LDT];

  const int tid = threadIdx.x;
  const int lane = tid & 31, w = tid >> 5;
  const int rowBase = blockIdx.y * BM, nBase = blockIdx.x * BN;
  const int nOff = (w & 3) * 16, mOff = (w >> 2) * 32;

  // cooperative tile staging: 256 threads, 16B each (row = tid>>2, chunk = tid&3)
  const int lr = tid >> 2, lc = tid & 3;
  const __bf16* ag = X + (size_t)(rowBase + lr) * Kdim + lc * 8;
  const __bf16* gg = Wgt + (size_t)(nBase + lr) * Kdim + lc * 8;
  const __bf16* ug = Wut + (size_t)(nBase + lr) * Kdim + lc * 8;
  const int ldsOff = lr * LDT + lc * 8;
  const unsigned aL[2] = {lds_lo32(&As[0][ldsOff]), lds_lo32(&As[1][ldsOff])};
  const unsigned gL[2] = {lds_lo32(&Bg[0][ldsOff]), lds_lo32(&Bg[1][ldsOff])};
  const unsigned uL[2] = {lds_lo32(&Bu[0][ldsOff]), lds_lo32(&Bu[1][ldsOff])};

  v8f cg0 = {}, cg1 = {}, cu0 = {}, cu1 = {};

  // preload tile 0 via async DMA (global -> LDS, no VGPR staging)
  async_copy_b128(aL[0], ag);
  async_copy_b128(gL[0], gg);
  async_copy_b128(uL[0], ug);

  const int nk = Kdim >> 5;
  const int khalf = lane >> 4;
  const int fr = lane & 15;

  for (int kk = 0; kk < nk; ++kk) {
    const int cur = kk & 1, nxt = cur ^ 1;
    wait_async0();    // my async writes into buf[cur] are complete
    __syncthreads();  // everyone's are; nobody still reads buf[nxt]
    if (kk + 1 < nk) {
      async_copy_b128(aL[nxt], ag + (kk + 1) * 32);
      async_copy_b128(gL[nxt], gg + (kk + 1) * 32);
      async_copy_b128(uL[nxt], ug + (kk + 1) * 32);
    }
    // fragments (documented CDNA5 WMMA VGPR layouts)
    v16bf a0 = ld_frag_split(&As[cur][(mOff + fr) * LDT + khalf * 8]);
    v16bf a1 = ld_frag_split(&As[cur][(mOff + 16 + fr) * LDT + khalf * 8]);
    v16bf bg = ld_frag_split(&Bg[cur][(nOff + fr) * LDT + khalf * 8]);
    v16bf bu = ld_frag_split(&Bu[cur][(nOff + fr) * LDT + khalf * 8]);

    cg0 = wmma_bf16(a0, bg, cg0);
    cg1 = wmma_bf16(a1, bg, cg1);
    cu0 = wmma_bf16(a0, bu, cu0);
    cu1 = wmma_bf16(a1, bu, cu1);
  }

  // epilogue: h = silu(g) * u  (C/D layout: vgpr r -> M=r+8*(lane>=16), N=lane&15)
  const int nG = nBase + nOff + fr;
  const int mHalf = khalf * 8;
#pragma unroll
  for (int i = 0; i < 8; ++i) {
    int m0 = rowBase + mOff + mHalf + i;
    Hact[(size_t)m0 * Ndim + nG] = (__bf16)(silu(cg0[i]) * cu0[i]);
    int m1 = m0 + 16;
    Hact[(size_t)m1 * Ndim + nG] = (__bf16)(silu(cg1[i]) * cu1[i]);
  }
}

// ------------------- GEMM2: Y = Hact@Wd, scaled accumulate -----------------
// Hact:[T,K] bf16; Wdt:[N,K] bf16; Out:[T,N] fp32.
// comb != nullptr: Out += comb[row,e] * Y   (sequential launches -> no atomics)
// comb == nullptr: Out  = Y                 (shared expert runs first)

__global__ __launch_bounds__(256) void gemm2_down_kernel(
    const __bf16* __restrict__ Hact, const __bf16* __restrict__ Wdt,
    float* __restrict__ Out, const float* __restrict__ comb, int eIdx,
    int Kdim, int Ndim) {
  __shared__ __bf16 As[2][BM * LDT];
  __shared__ __bf16 Bs[2][BN * LDT];

  const int tid = threadIdx.x;
  const int lane = tid & 31, w = tid >> 5;
  const int rowBase = blockIdx.y * BM, nBase = blockIdx.x * BN;
  const int nOff = (w & 3) * 16, mOff = (w >> 2) * 32;

  const int lr = tid >> 2, lc = tid & 3;
  const __bf16* ag = Hact + (size_t)(rowBase + lr) * Kdim + lc * 8;
  const __bf16* bgp = Wdt + (size_t)(nBase + lr) * Kdim + lc * 8;
  const int ldsOff = lr * LDT + lc * 8;
  const unsigned aL[2] = {lds_lo32(&As[0][ldsOff]), lds_lo32(&As[1][ldsOff])};
  const unsigned bL[2] = {lds_lo32(&Bs[0][ldsOff]), lds_lo32(&Bs[1][ldsOff])};

  v8f c0 = {}, c1 = {};

  async_copy_b128(aL[0], ag);
  async_copy_b128(bL[0], bgp);

  const int nk = Kdim >> 5;
  const int khalf = lane >> 4;
  const int fr = lane & 15;

  for (int kk = 0; kk < nk; ++kk) {
    const int cur = kk & 1, nxt = cur ^ 1;
    wait_async0();
    __syncthreads();
    if (kk + 1 < nk) {
      async_copy_b128(aL[nxt], ag + (kk + 1) * 32);
      async_copy_b128(bL[nxt], bgp + (kk + 1) * 32);
    }
    v16bf a0 = ld_frag_split(&As[cur][(mOff + fr) * LDT + khalf * 8]);
    v16bf a1 = ld_frag_split(&As[cur][(mOff + 16 + fr) * LDT + khalf * 8]);
    v16bf b0 = ld_frag_split(&Bs[cur][(nOff + fr) * LDT + khalf * 8]);

    c0 = wmma_bf16(a0, b0, c0);
    c1 = wmma_bf16(a1, b0, c1);
  }

  const int nG = nBase + nOff + fr;
  const int mHalf = khalf * 8;
#pragma unroll
  for (int i = 0; i < 8; ++i) {
    int m0 = rowBase + mOff + mHalf + i;
    int m1 = m0 + 16;
    size_t o0 = (size_t)m0 * Ndim + nG;
    size_t o1 = (size_t)m1 * Ndim + nG;
    if (comb) {
      Out[o0] += comb[(size_t)m0 * NEXP + eIdx] * c0[i];
      Out[o1] += comb[(size_t)m1 * NEXP + eIdx] * c1[i];
    } else {
      Out[o0] = c0[i];
      Out[o1] = c1[i];
    }
  }
}

// ------------------------------- launcher ----------------------------------

extern "C" void kernel_launch(void* const* d_in, const int* in_sizes, int n_in,
                              void* d_out, int out_size, void* d_ws,
                              size_t ws_size, hipStream_t stream) {
  const float* hs       = (const float*)d_in[0];  // [B,S,D]
  const float* router_w = (const float*)d_in[1];  // [D,E]
  const float* w_gate   = (const float*)d_in[2];  // [E,D,H]
  const float* w_up     = (const float*)d_in[3];  // [E,D,H]
  const float* w_down   = (const float*)d_in[4];  // [E,H,D]
  const float* sw_gate  = (const float*)d_in[5];  // [D,HS]
  const float* sw_up    = (const float*)d_in[6];  // [D,HS]
  const float* sw_down  = (const float*)d_in[7];  // [HS,D]
  float* out = (float*)d_out;                     // [B,S,D]

  // workspace carve-out (~160 MB total)
  char* ws = (char*)d_ws;
  size_t off = 0;
  auto carve = [&](size_t bytes) {
    void* p = ws + off;
    off = (off + bytes + 255) & ~(size_t)255;
    return p;
  };
  __bf16* Xb   = (__bf16*)carve((size_t)T_TOK * DIM * 2);
  __bf16* Wgt  = (__bf16*)carve((size_t)NEXP * HDIM * DIM * 2);
  __bf16* Wut  = (__bf16*)carve((size_t)NEXP * HDIM * DIM * 2);
  __bf16* Wdt  = (__bf16*)carve((size_t)NEXP * DIM * HDIM * 2);
  __bf16* SWgt = (__bf16*)carve((size_t)HSDIM * DIM * 2);
  __bf16* SWut = (__bf16*)carve((size_t)HSDIM * DIM * 2);
  __bf16* SWdt = (__bf16*)carve((size_t)DIM * HSDIM * 2);
  float*  comb = (float*)carve((size_t)T_TOK * NEXP * 4);
  __bf16* Hact = (__bf16*)carve((size_t)T_TOK * HSDIM * 2);  // max(H,HS)

  // ---- precision/layout conversion (fp32 -> bf16, weights transposed) ----
  {
    long long n = (long long)T_TOK * DIM;
    cvt_bf16_kernel<<<(unsigned)((n / 4 + 255) / 256), 256, 0, stream>>>(hs, Xb, n);
  }
  transpose_cvt_kernel<<<dim3(HDIM / 32, DIM / 32, NEXP), dim3(32, 8), 0, stream>>>(
      w_gate, Wgt, DIM, HDIM);
  transpose_cvt_kernel<<<dim3(HDIM / 32, DIM / 32, NEXP), dim3(32, 8), 0, stream>>>(
      w_up, Wut, DIM, HDIM);
  transpose_cvt_kernel<<<dim3(DIM / 32, HDIM / 32, NEXP), dim3(32, 8), 0, stream>>>(
      w_down, Wdt, HDIM, DIM);
  transpose_cvt_kernel<<<dim3(HSDIM / 32, DIM / 32, 1), dim3(32, 8), 0, stream>>>(
      sw_gate, SWgt, DIM, HSDIM);
  transpose_cvt_kernel<<<dim3(HSDIM / 32, DIM / 32, 1), dim3(32, 8), 0, stream>>>(
      sw_up, SWut, DIM, HSDIM);
  transpose_cvt_kernel<<<dim3(DIM / 32, HSDIM / 32, 1), dim3(32, 8), 0, stream>>>(
      sw_down, SWdt, HSDIM, DIM);

  // ---- router: combine weights [T, E] ----
  router_kernel<<<T_TOK, 256, 0, stream>>>(hs, router_w, comb);

  // ---- shared expert (writes out) ----
  gemm1_swiglu_kernel<<<dim3(HSDIM / BN, T_TOK / BM), 256, 0, stream>>>(
      Xb, SWgt, SWut, Hact, DIM, HSDIM);
  gemm2_down_kernel<<<dim3(DIM / BN, T_TOK / BM), 256, 0, stream>>>(
      Hact, SWdt, out, nullptr, 0, HSDIM, DIM);

  // ---- routed experts (accumulate into out, weighted by combine) ----
  for (int e = 0; e < NEXP; ++e) {
    gemm1_swiglu_kernel<<<dim3(HDIM / BN, T_TOK / BM), 256, 0, stream>>>(
        Xb, Wgt + (size_t)e * HDIM * DIM, Wut + (size_t)e * HDIM * DIM, Hact,
        DIM, HDIM);
    gemm2_down_kernel<<<dim3(DIM / BN, T_TOK / BM), 256, 0, stream>>>(
        Hact, Wdt + (size_t)e * DIM * HDIM, out, comb, e, HDIM, DIM);
  }
}
